// TernaryMHA_59562606461171
// MI455X (gfx1250) — matrix operand order repaired and checked by reading, not compiled, not run on hardware
//
#include <hip/hip_runtime.h>

#define DEV __device__ __forceinline__

typedef __attribute__((ext_vector_type(16))) __bf16 v16bf;
typedef __attribute__((ext_vector_type(8)))  float  v8f;

union FragBF { v16bf v; unsigned short u[16]; };

DEV unsigned short f2bf(float f) {
  unsigned int u; __builtin_memcpy(&u, &f, 4);
  unsigned int r = u + 0x7fffu + ((u >> 16) & 1u);   // round-to-nearest-even
  return (unsigned short)(r >> 16);
}
DEV float bf2f(unsigned short h) {
  unsigned int u = ((unsigned int)h) << 16;
  float f; __builtin_memcpy(&f, &u, 4); return f;
}

DEV v8f wmma_bf16(v16bf a, v16bf b, v8f c) {
  // 8 args: (neg_a, A, neg_b, B, c_mod, C, reuse_a, reuse_b)
  return __builtin_amdgcn_wmma_f32_16x16x32_bf16(false, a, false, b, (short)0, c, false, false);
}

// A-fragment K index for element j, lane-half h (per CDNA5 ISA 16-bit A 16x32 layout)
DEV int kmapA(int j, int h) { return 8 * h + j + ((j >= 8) ? 8 : 0); }

// --- CDNA5 async global->LDS copy (ASYNCcnt-tracked), GV addressing mode ----
// per-lane: LDS[vdst] = MEM[vaddr], 16 bytes
DEV void async_copy_b128(unsigned lds_off, const void* gaddr) {
  asm volatile("global_load_async_to_lds_b128 %0, %1, off"
               :: "v"(lds_off), "v"(gaddr) : "memory");
}
template <int N>
DEV void wait_asynccnt() {
  asm volatile("s_wait_asynccnt %0" :: "n"(N) : "memory");
}
// LDS byte offset of a __shared__ object (generic-pointer low 32 bits)
DEV unsigned lds_off_of(const void* p) { return (unsigned)(size_t)p; }

// ---------------------------------------------------------------------------
// Weight preprocessing: ternarization statistics (deterministic 2-pass)
// ---------------------------------------------------------------------------
__global__ __launch_bounds__(256) void k_absum(const float* __restrict__ W,
                                               float* __restrict__ part) {
  __shared__ float sm[256];
  const int tid = threadIdx.x;
  const size_t base = (size_t)blockIdx.x * 1024;
  float s = 0.f;
#pragma unroll
  for (int j = 0; j < 4; ++j) s += fabsf(W[base + tid + j * 256]);
  sm[tid] = s; __syncthreads();
  for (int st = 128; st > 0; st >>= 1) { if (tid < st) sm[tid] += sm[tid + st]; __syncthreads(); }
  if (tid == 0) part[blockIdx.x] = sm[0];
}

__global__ __launch_bounds__(256) void k_delta(const float* __restrict__ part,
                                               float* __restrict__ stats) {
  __shared__ float sm[256];
  const int tid = threadIdx.x;
  float s = 0.f;
  for (int i = tid; i < 1024; i += 256) s += part[i];
  sm[tid] = s; __syncthreads();
  for (int st = 128; st > 0; st >>= 1) { if (tid < st) sm[tid] += sm[tid + st]; __syncthreads(); }
  if (tid == 0) stats[0] = 0.05f * sm[0] / 1048576.0f;   // THRESH * mean|W|
}

__global__ __launch_bounds__(256) void k_ternarize(const float* __restrict__ W,
                                                   const float* __restrict__ stats,
                                                   float* __restrict__ part2,
                                                   unsigned short* __restrict__ wt) {
  __shared__ float s1[256], s2[256];
  const float delta = stats[0];
  const int tid = threadIdx.x;
  const size_t base = (size_t)blockIdx.x * 1024;
  float a1 = 0.f, a2 = 0.f;
#pragma unroll
  for (int j = 0; j < 4; ++j) {
    size_t i = base + tid + j * 256;
    float w = W[i], a = fabsf(w);
    bool m = a > delta;
    a1 += m ? a : 0.f;
    a2 += m ? 1.f : 0.f;
    wt[i] = (unsigned short)(m ? (w < 0.f ? 0xBF80u : 0x3F80u) : 0u);  // bf16 {-1,0,+1}
  }
  s1[tid] = a1; s2[tid] = a2; __syncthreads();
  for (int st = 128; st > 0; st >>= 1) {
    if (tid < st) { s1[tid] += s1[tid + st]; s2[tid] += s2[tid + st]; }
    __syncthreads();
  }
  if (tid == 0) { part2[2 * blockIdx.x] = s1[0]; part2[2 * blockIdx.x + 1] = s2[0]; }
}

__global__ __launch_bounds__(256) void k_alpha(const float* __restrict__ part2,
                                               float* __restrict__ stats) {
  __shared__ float s1[256], s2[256];
  const int tid = threadIdx.x;
  float a1 = 0.f, a2 = 0.f;
  for (int i = tid; i < 1024; i += 256) { a1 += part2[2 * i]; a2 += part2[2 * i + 1]; }
  s1[tid] = a1; s2[tid] = a2; __syncthreads();
  for (int st = 128; st > 0; st >>= 1) {
    if (tid < st) { s1[tid] += s1[tid + st]; s2[tid] += s2[tid + st]; }
    __syncthreads();
  }
  if (tid == 0) stats[1] = s1[0] / fmaxf(s2[0], 1.0f);
}

__global__ __launch_bounds__(256) void k_cast_bf16(const float* __restrict__ x,
                                                   unsigned short* __restrict__ xb, int n) {
  int i = blockIdx.x * 256 + threadIdx.x;
  if (i < n) xb[i] = f2bf(x[i]);
}

// ---------------------------------------------------------------------------
// Projection GEMM (NT): y[m,o] = alpha * sum_c xb[m,c] * wt[o,c] + bias[o]
// 64x64 block tile, 4 waves x (2x2) WMMA tiles, K-step 32.
// Double-buffered LDS, staged with global_load_async_to_lds_b128 (ASYNCcnt).
// ---------------------------------------------------------------------------
__global__ __launch_bounds__(128) void k_proj_qkv(const unsigned short* __restrict__ xb,
                                                  const unsigned short* __restrict__ wt,
                                                  const float* __restrict__ stats,
                                                  const float* __restrict__ bias,
                                                  unsigned short* __restrict__ outHS) {
  __shared__ unsigned short ldsA[2][64 * 32];
  __shared__ unsigned short ldsB[2][64 * 32];
  const int tid = threadIdx.x, lane = tid & 31, wave = tid >> 5;
  const int wm = wave >> 1, wn = wave & 1;
  const int m0 = blockIdx.x * 64, n0 = blockIdx.y * 64;
  const int ln = lane & 15, hh = lane >> 4;
  v8f zero = {};
  v8f acc[2][2];
#pragma unroll
  for (int i = 0; i < 2; ++i)
#pragma unroll
    for (int j = 0; j < 2; ++j) acc[i][j] = zero;

  const int lrow = tid >> 1, lseg = (tid & 1) * 16;
  const unsigned short* srcA = xb + (size_t)(m0 + lrow) * 1024 + lseg;
  const unsigned short* srcB = wt + (size_t)(n0 + lrow) * 1024 + lseg;

  auto stage = [&](int k0, int buf) {
    unsigned ao = lds_off_of(&ldsA[buf][lrow * 32 + lseg]);
    unsigned bo = lds_off_of(&ldsB[buf][lrow * 32 + lseg]);
    async_copy_b128(ao,      srcA + k0);
    async_copy_b128(ao + 16, srcA + k0 + 8);
    async_copy_b128(bo,      srcB + k0);
    async_copy_b128(bo + 16, srcB + k0 + 8);
  };

  stage(0, 0);
  int cur = 0;
  for (int k0 = 0; k0 < 1024; k0 += 32) {
    if (k0 + 32 < 1024) {
      stage(k0 + 32, cur ^ 1);   // prefetch next tile into other buffer
      wait_asynccnt<4>();        // wait only for the 4 older (current-tile) copies
    } else {
      wait_asynccnt<0>();
    }
    __syncthreads();

    FragBF a[2], b[2];
#pragma unroll
    for (int mt = 0; mt < 2; ++mt) {
      int base = (wm * 32 + mt * 16 + ln) * 32;
#pragma unroll
      for (int j = 0; j < 16; ++j) a[mt].u[j] = ldsA[cur][base + 16 * hh + j];
    }
#pragma unroll
    for (int nt = 0; nt < 2; ++nt) {
      int base = (wn * 32 + nt * 16 + ln) * 32 + 16 * hh;
#pragma unroll
      for (int j = 0; j < 16; ++j) b[nt].u[j] = ldsB[cur][base + j];
    }
#pragma unroll
    for (int mt = 0; mt < 2; ++mt)
#pragma unroll
      for (int nt = 0; nt < 2; ++nt)
        acc[mt][nt] = wmma_bf16(a[mt].v, b[nt].v, acc[mt][nt]);
    __syncthreads();             // all waves done reading `cur` before it is restaged
    cur ^= 1;
  }

  const float alpha = stats[1];
#pragma unroll
  for (int mt = 0; mt < 2; ++mt)
#pragma unroll
    for (int nt = 0; nt < 2; ++nt)
#pragma unroll
      for (int r = 0; r < 8; ++r) {
        int row = m0 + wm * 32 + mt * 16 + r + 8 * hh;   // D: M = r + 8h
        int col = n0 + wn * 32 + nt * 16 + ln;           // D: N = lane&15
        float val = acc[mt][nt][r] * alpha + bias[col];
        int bb = row >> 11, t = row & 2047, head = col >> 6, d = col & 63;
        outHS[(((size_t)(bb * 16 + head) * 2048 + t) << 6) + d] = f2bf(val);
      }
}

// Output projection: same mainloop, fp32 output + |y| max partials for fp4 quant
__global__ __launch_bounds__(128) void k_proj_out(const unsigned short* __restrict__ yb,
                                                  const unsigned short* __restrict__ wt,
                                                  const float* __restrict__ stats,
                                                  const float* __restrict__ bias,
                                                  float* __restrict__ out,
                                                  float* __restrict__ maxpart) {
  __shared__ unsigned short ldsA[2][64 * 32];
  __shared__ unsigned short ldsB[2][64 * 32];
  __shared__ float red[128];
  const int tid = threadIdx.x, lane = tid & 31, wave = tid >> 5;
  const int wm = wave >> 1, wn = wave & 1;
  const int m0 = blockIdx.x * 64, n0 = blockIdx.y * 64;
  const int ln = lane & 15, hh = lane >> 4;
  v8f zero = {};
  v8f acc[2][2];
#pragma unroll
  for (int i = 0; i < 2; ++i)
#pragma unroll
    for (int j = 0; j < 2; ++j) acc[i][j] = zero;

  const int lrow = tid >> 1, lseg = (tid & 1) * 16;
  const unsigned short* srcA = yb + (size_t)(m0 + lrow) * 1024 + lseg;
  const unsigned short* srcB = wt + (size_t)(n0 + lrow) * 1024 + lseg;

  auto stage = [&](int k0, int buf) {
    unsigned ao = lds_off_of(&ldsA[buf][lrow * 32 + lseg]);
    unsigned bo = lds_off_of(&ldsB[buf][lrow * 32 + lseg]);
    async_copy_b128(ao,      srcA + k0);
    async_copy_b128(ao + 16, srcA + k0 + 8);
    async_copy_b128(bo,      srcB + k0);
    async_copy_b128(bo + 16, srcB + k0 + 8);
  };

  stage(0, 0);
  int cur = 0;
  for (int k0 = 0; k0 < 1024; k0 += 32) {
    if (k0 + 32 < 1024) { stage(k0 + 32, cur ^ 1); wait_asynccnt<4>(); }
    else                { wait_asynccnt<0>(); }
    __syncthreads();

    FragBF a[2], b[2];
#pragma unroll
    for (int mt = 0; mt < 2; ++mt) {
      int base = (wm * 32 + mt * 16 + ln) * 32;
#pragma unroll
      for (int j = 0; j < 16; ++j) a[mt].u[j] = ldsA[cur][base + 16 * hh + j];
    }
#pragma unroll
    for (int nt = 0; nt < 2; ++nt) {
      int base = (wn * 32 + nt * 16 + ln) * 32 + 16 * hh;
#pragma unroll
      for (int j = 0; j < 16; ++j) b[nt].u[j] = ldsB[cur][base + j];
    }
#pragma unroll
    for (int mt = 0; mt < 2; ++mt)
#pragma unroll
      for (int nt = 0; nt < 2; ++nt)
        acc[mt][nt] = wmma_bf16(a[mt].v, b[nt].v, acc[mt][nt]);
    __syncthreads();
    cur ^= 1;
  }

  const float alpha = stats[1];
  float amax = 0.f;
#pragma unroll
  for (int mt = 0; mt < 2; ++mt)
#pragma unroll
    for (int nt = 0; nt < 2; ++nt)
#pragma unroll
      for (int r = 0; r < 8; ++r) {
        int row = m0 + wm * 32 + mt * 16 + r + 8 * hh;
        int col = n0 + wn * 32 + nt * 16 + ln;
        float val = acc[mt][nt][r] * alpha + bias[col];
        out[(size_t)row * 1024 + col] = val;
        amax = fmaxf(amax, fabsf(val));
      }
  red[tid] = amax; __syncthreads();
  for (int st = 64; st > 0; st >>= 1) { if (tid < st) red[tid] = fmaxf(red[tid], red[tid + st]); __syncthreads(); }
  if (tid == 0) maxpart[blockIdx.y * 64 + blockIdx.x] = red[0];
}

// ---------------------------------------------------------------------------
// Attention: one wave per (b, h, 16-row q-tile). Scores (16 x 2048) in LDS
// as bf16. QK^T and P@V via WMMA; tiles staged with async global->LDS copies.
// ---------------------------------------------------------------------------
__global__ __launch_bounds__(32) void k_attn(const unsigned short* __restrict__ qb,
                                             const unsigned short* __restrict__ kb,
                                             const unsigned short* __restrict__ vb,
                                             unsigned short* __restrict__ yb) {
  __shared__ unsigned short sS[16 * 2048];  // 64 KB scores/probabilities
  __shared__ unsigned short sT[32 * 64];    // K/V staging tile
  __shared__ unsigned short sQ[16 * 64];    // Q tile
  const int lane = threadIdx.x, ln = lane & 15, hh = lane >> 4;
  const int blk = blockIdx.x, t0 = (blk & 127) * 16, bh = blk >> 7;

  const unsigned short* qp = qb + (size_t)bh * 2048 * 64 + (size_t)t0 * 64;
  { int row = lane >> 1, off = (lane & 1) * 32;
    unsigned qo = lds_off_of(&sQ[row * 64 + off]);
    const unsigned short* p = qp + row * 64 + off;
#pragma unroll
    for (int i = 0; i < 4; ++i) async_copy_b128(qo + 16 * i, p + 8 * i);
  }
  wait_asynccnt<0>();
  __syncthreads();

  FragBF aq[2];
#pragma unroll
  for (int kk = 0; kk < 2; ++kk)
#pragma unroll
    for (int j = 0; j < 16; ++j) aq[kk].u[j] = sQ[ln * 64 + 32 * kk + 16 * hh + j];

  // S = (Q K^T) / sqrt(Dh)
  const unsigned short* kp = kb + (size_t)bh * 2048 * 64;
  for (int st = 0; st < 128; ++st) {
    int s0 = st * 16;
    { int row = lane >> 1, off = (lane & 1) * 32;
      unsigned to = lds_off_of(&sT[row * 64 + off]);
      const unsigned short* p = kp + (size_t)(s0 + row) * 64 + off;
#pragma unroll
      for (int i = 0; i < 4; ++i) async_copy_b128(to + 16 * i, p + 8 * i);
    }
    wait_asynccnt<0>();
    __syncthreads();
    v8f c = {};
#pragma unroll
    for (int kk = 0; kk < 2; ++kk) {
      FragBF bk;
#pragma unroll
      for (int j = 0; j < 16; ++j) bk.u[j] = sT[ln * 64 + 32 * kk + 16 * hh + j];
      c = wmma_bf16(aq[kk].v, bk.v, c);
    }
#pragma unroll
    for (int r = 0; r < 8; ++r)
      sS[(r + 8 * hh) * 2048 + s0 + ln] = f2bf(c[r] * 0.125f);   // 1/sqrt(64)
    __syncthreads();
  }

  // softmax: lanes (ln, ln+16) split row ln's 2048 columns in half
  const int base = ln * 2048 + hh * 1024;
  float mx = -3.0e38f;
  for (int i = 0; i < 1024; ++i) mx = fmaxf(mx, bf2f(sS[base + i]));
  mx = fmaxf(mx, __shfl_xor(mx, 16, 32));
  float sum = 0.f;
  for (int i = 0; i < 1024; ++i) {
    float e = __expf(bf2f(sS[base + i]) - mx);
    sS[base + i] = f2bf(e);
    sum += e;
  }
  sum += __shfl_xor(sum, 16, 32);
  const float inv = 1.0f / sum;
  __syncthreads();

  // O = P @ V  (M=16, N=64 => 4 accumulators, K-step 32)
  const unsigned short* vp = vb + (size_t)bh * 2048 * 64;
  v8f zero = {};
  v8f co[4];
#pragma unroll
  for (int nt = 0; nt < 4; ++nt) co[nt] = zero;
  for (int s0 = 0; s0 < 2048; s0 += 32) {
    { unsigned to = lds_off_of(&sT[lane * 64]);
      const unsigned short* p = vp + (size_t)(s0 + lane) * 64;
#pragma unroll
      for (int q = 0; q < 8; ++q) async_copy_b128(to + 16 * q, p + 8 * q);
    }
    wait_asynccnt<0>();
    __syncthreads();
    FragBF ap;
#pragma unroll
    for (int j = 0; j < 16; ++j)
      ap.u[j] = f2bf(bf2f(sS[ln * 2048 + s0 + 16 * hh + j]) * inv);
#pragma unroll
    for (int nt = 0; nt < 4; ++nt) {
      FragBF bv;
#pragma unroll
      for (int j = 0; j < 16; ++j) bv.u[j] = sT[(16 * hh + j) * 64 + nt * 16 + ln];
      co[nt] = wmma_bf16(ap.v, bv.v, co[nt]);
    }
    __syncthreads();
  }

  const int bb = bh >> 4, head = bh & 15;
#pragma unroll
  for (int nt = 0; nt < 4; ++nt)
#pragma unroll
    for (int r = 0; r < 8; ++r) {
      int t = t0 + r + 8 * hh, d = nt * 16 + ln;
      yb[(size_t)(bb * 2048 + t) * 1024 + head * 64 + d] = f2bf(co[nt][r]);
    }
}

// ---------------------------------------------------------------------------
// fp4 activation quant: scale = max(max|y|/7, 1e-8); y = clip(round(y/s),±7)*s
// ---------------------------------------------------------------------------
__global__ __launch_bounds__(256) void k_qscale(const float* __restrict__ maxpart,
                                                float* __restrict__ qs) {
  __shared__ float sm[256];
  const int tid = threadIdx.x;
  float m = 0.f;
  for (int i = tid; i < 1024; i += 256) m = fmaxf(m, maxpart[i]);
  sm[tid] = m; __syncthreads();
  for (int st = 128; st > 0; st >>= 1) { if (tid < st) sm[tid] = fmaxf(sm[tid], sm[tid + st]); __syncthreads(); }
  if (tid == 0) qs[0] = fmaxf(sm[0] / 7.0f, 1e-8f);
}

__global__ __launch_bounds__(256) void k_quant(float* __restrict__ y,
                                               const float* __restrict__ qs, int n) {
  int i = blockIdx.x * 256 + threadIdx.x;
  if (i < n) {
    float s = qs[0];
    float q = rintf(y[i] / s);
    q = fminf(fmaxf(q, -7.f), 7.f);
    y[i] = q * s;
  }
}

// ---------------------------------------------------------------------------
extern "C" void kernel_launch(void* const* d_in, const int* in_sizes, int n_in,
                              void* d_out, int out_size, void* d_ws, size_t ws_size,
                              hipStream_t stream) {
  const float* x  = (const float*)d_in[0];
  const float* Wq = (const float*)d_in[1];
  const float* bq = (const float*)d_in[2];
  const float* Wk = (const float*)d_in[3];
  const float* bk = (const float*)d_in[4];
  const float* Wv = (const float*)d_in[5];
  const float* bv = (const float*)d_in[6];
  const float* Wo = (const float*)d_in[7];
  const float* bo = (const float*)d_in[8];
  float* out = (float*)d_out;

  char* ws = (char*)d_ws;
  size_t off = 0;
  auto take = [&](size_t bytes) -> char* {
    char* p = ws + off;
    off = (off + bytes + 255) & ~(size_t)255;
    return p;
  };
  unsigned short* xb = (unsigned short*)take((size_t)4096 * 1024 * 2);
  unsigned short* wt[4];
  for (int i = 0; i < 4; ++i) wt[i] = (unsigned short*)take((size_t)1024 * 1024 * 2);
  unsigned short* qbuf = (unsigned short*)take((size_t)4096 * 1024 * 2);
  unsigned short* kbuf = (unsigned short*)take((size_t)4096 * 1024 * 2);
  unsigned short* vbuf = (unsigned short*)take((size_t)4096 * 1024 * 2);
  unsigned short* ybuf = (unsigned short*)take((size_t)4096 * 1024 * 2);
  float* part    = (float*)take(1024 * 4);
  float* part2   = (float*)take(2048 * 4);
  float* statsW[4];
  for (int i = 0; i < 4; ++i) statsW[i] = (float*)take(256);
  float* maxpart = (float*)take(1024 * 4);
  float* qs      = (float*)take(256);

  const float* Ws[4] = {Wq, Wk, Wv, Wo};
  for (int i = 0; i < 4; ++i) {
    k_absum<<<1024, 256, 0, stream>>>(Ws[i], part);
    k_delta<<<1, 256, 0, stream>>>(part, statsW[i]);
    k_ternarize<<<1024, 256, 0, stream>>>(Ws[i], statsW[i], part2, wt[i]);
    k_alpha<<<1, 256, 0, stream>>>(part2, statsW[i]);
  }
  const int NX = 4096 * 1024;
  k_cast_bf16<<<(NX + 255) / 256, 256, 0, stream>>>(x, xb, NX);

  dim3 g(64, 16), blk(128);
  k_proj_qkv<<<g, blk, 0, stream>>>(xb, wt[0], statsW[0], bq, qbuf);
  k_proj_qkv<<<g, blk, 0, stream>>>(xb, wt[1], statsW[1], bk, kbuf);
  k_proj_qkv<<<g, blk, 0, stream>>>(xb, wt[2], statsW[2], bv, vbuf);

  k_attn<<<4096, 32, 0, stream>>>(qbuf, kbuf, vbuf, ybuf);

  k_proj_out<<<g, blk, 0, stream>>>(ybuf, wt[3], statsW[3], bo, out, maxpart);
  k_qscale<<<1, 256, 0, stream>>>(maxpart, qs);
  k_quant<<<(NX + 255) / 256, 256, 0, stream>>>(out, qs, NX);

  (void)in_sizes; (void)n_in; (void)out_size; (void)ws_size;
}
